// Receiver_6399501271456
// MI455X (gfx1250) — compile-verified, matrix operand order checked by reading, not run on hardware
//
#include <hip/hip_runtime.h>
#include <hip/hip_bf16.h>
#include <math.h>
#include <stdint.h>

typedef __bf16 bf16;
typedef __bf16 bf16x16 __attribute__((ext_vector_type(16)));
typedef float  f32x8   __attribute__((ext_vector_type(8)));

#define BDIM   256
#define BM     128
#define BN     128
#define BK     32
#define APITCH 40     // bf16 per LDS row: 32 data + 8 pad (20 banks -> conflict free)
#define GPITCH 129    // f32 staging pitch (odd -> conflict free)

#define B_ 512
#define T_ 64
#define V_ 1024
#define E_ 256
#define H_ 512
#define O_ 256

// ---- CDNA5 async Global->LDS (ASYNCcnt path) -------------------------------
// LDS destination address = low 32 bits of the flat shared pointer
// (ISA 10.2: LDS aperture maps addr[31:0] to the LDS address).
__device__ __forceinline__ void async_b128(unsigned lds_addr, const void* gptr)
{
    asm volatile("global_load_async_to_lds_b128 %0, %1, off"
                 :: "v"(lds_addr), "v"((unsigned long long)(uintptr_t)gptr)
                 : "memory");
}
__device__ __forceinline__ void wait_async0()
{
    asm volatile("s_wait_asynccnt 0x0" ::: "memory");
}
__device__ __forceinline__ unsigned lds_addr_of(const void* p)
{
    return (unsigned)(uintptr_t)p;
}

// issue one 128x32 bf16 tile: 256 threads x 32B, two async b128 per thread
__device__ __forceinline__ void async_tile(const bf16* src, size_t stride, bf16* Ls, int tid)
{
    const int row = tid >> 1, c0 = (tid & 1) * 16;
    const bf16* g = src + (size_t)row * stride + c0;
    const unsigned l = lds_addr_of(Ls + row * APITCH + c0);
    async_b128(l,      g);
    async_b128(l + 16, (const char*)g + 16);
}

// ---- fragment gather from LDS, per documented 16-bit A/B layout -------------
// lane = kh*16 + idx16 ; VGPR r holds K pair: K = (r<4 ? 2r : 16+2(r-4)) + 8*kh
__device__ __forceinline__ void frag_load(const bf16* __restrict__ Ls, int row0,
                                          int lane, bf16x16* frag)
{
    const int i16 = lane & 15, kh = lane >> 4;
    const unsigned int* row = (const unsigned int*)(Ls + (row0 + i16) * APITCH);
    unsigned int* fv = (unsigned int*)frag;
#pragma unroll
    for (int r = 0; r < 8; ++r) {
        const int k = ((r < 4) ? (2 * r) : (16 + 2 * (r - 4))) + kh * 8;
        fv[r] = row[k >> 1];
    }
}

// 8 WMMA tiles per wave: 32 rows (wr) x 64 cols (wc) of the 128x128 block tile
__device__ __forceinline__ void mma_8(const bf16* As, const bf16* Bs,
                                      f32x8 acc[2][4], int wr, int wc, int lane)
{
    bf16x16 a0, a1, b[4];
    frag_load(As, wr * 32,      lane, &a0);
    frag_load(As, wr * 32 + 16, lane, &a1);
#pragma unroll
    for (int j = 0; j < 4; ++j) frag_load(Bs, wc * 64 + j * 16, lane, &b[j]);
#pragma unroll
    for (int j = 0; j < 4; ++j) {
        acc[0][j] = __builtin_amdgcn_wmma_f32_16x16x32_bf16(
            false, a0, false, b[j], (short)0, acc[0][j], false, false);
        acc[1][j] = __builtin_amdgcn_wmma_f32_16x16x32_bf16(
            false, a1, false, b[j], (short)0, acc[1][j], false, false);
    }
}

// ---- f32 tile staging (needs cvt -> cannot use async DMA) -------------------
struct F32Tile { float4 x, y, z, w; };
__device__ __forceinline__ F32Tile fetch_f32tile(const float* src, size_t stride, int tid)
{
    const int row = tid >> 1, c0 = (tid & 1) * 16;
    const float4* s = (const float4*)(src + (size_t)row * stride + c0);
    F32Tile t;
    t.x = s[0]; t.y = s[1]; t.z = s[2]; t.w = s[3];
    return t;
}
__device__ __forceinline__ void commit_f32tile(const F32Tile& t, bf16* Ls, int tid)
{
    const int row = tid >> 1, c0 = (tid & 1) * 16;
    bf16* d = Ls + row * APITCH + c0;
    const float* f = (const float*)&t;
#pragma unroll
    for (int i = 0; i < 16; ++i) d[i] = (bf16)f[i];
}

__device__ __forceinline__ f32x8 f32x8_zero()
{
    f32x8 z = {0.f, 0.f, 0.f, 0.f, 0.f, 0.f, 0.f, 0.f};
    return z;
}

// ---- kernel 1: emb[t,b,:] = message[b,t,:] @ W_emb^T + b_emb  (bf16 out) ----
__global__ __launch_bounds__(BDIM) void embed_kernel(
    const float* __restrict__ msg,   // [B*T, V]  (m = b*T + t)
    const bf16*  __restrict__ Wemb,  // [E, V] bf16
    const float* __restrict__ bemb,  // [E]
    bf16*        __restrict__ emb)   // [T, B, E]
{
    __shared__ bf16 AsBuf[2][BM * APITCH];
    __shared__ bf16 BsBuf[2][BN * APITCH];
    const int tid = threadIdx.x, lane = tid & 31, wave = tid >> 5;
    const int wr = wave >> 1, wc = wave & 1;
    const int m0 = blockIdx.y * BM, n0 = blockIdx.x * BN;

    f32x8 acc[2][4];
#pragma unroll
    for (int i = 0; i < 2; ++i)
#pragma unroll
        for (int j = 0; j < 4; ++j) acc[i][j] = f32x8_zero();

    const int NS = V_ / BK;   // 32 stages
    // prologue: stage 0 (B via async DMA, A via VGPR staging + cvt)
    F32Tile ar = fetch_f32tile(msg + (size_t)m0 * V_, V_, tid);
    async_tile(Wemb + (size_t)n0 * V_, V_, BsBuf[0], tid);
    commit_f32tile(ar, AsBuf[0], tid);
    wait_async0();
    __syncthreads();

    for (int s = 0; s < NS; ++s) {
        const int nx = s + 1;
        if (nx < NS) {
            async_tile(Wemb + (size_t)n0 * V_ + nx * BK, V_, BsBuf[nx & 1], tid);
            ar = fetch_f32tile(msg + (size_t)m0 * V_ + nx * BK, V_, tid);
        }
        mma_8(AsBuf[s & 1], BsBuf[s & 1], acc, wr, wc, lane);
        if (nx < NS) {
            commit_f32tile(ar, AsBuf[nx & 1], tid);
            wait_async0();
        }
        __syncthreads();
    }

    const int kh = lane >> 4, n16 = lane & 15;
#pragma unroll
    for (int i = 0; i < 2; ++i)
#pragma unroll
        for (int j = 0; j < 4; ++j)
#pragma unroll
            for (int v = 0; v < 8; ++v) {
                const int m = m0 + wr * 32 + i * 16 + kh * 8 + v;
                const int n = n0 + wc * 64 + j * 16 + n16;
                const int t = m & (T_ - 1), b = m >> 6;
                emb[((size_t)t * B_ + b) * E_ + n] = (bf16)(acc[i][j][v] + bemb[n]);
            }
}

// ---- kernel 2: one LSTM step -----------------------------------------------
// Block covers 128 batch rows x 32 h-columns; its 128 N-cols are the 4 gate
// strips {gate*512 + hcol}, so i/f/g/o for each (b,hcol) stay in-block.
// Unified 24-stage pipeline: stages 0..7 = x-side (K=E), 8..23 = h-side (K=H).
__global__ __launch_bounds__(BDIM) void lstm_step_kernel(
    const bf16*  __restrict__ embt,   // [B, E]   (emb + t*B*E)
    const bf16*  __restrict__ Wih,    // [4H, E]
    const bf16*  __restrict__ Whh,    // [4H, H]
    const float* __restrict__ bih,    // [4H]
    const float* __restrict__ bhh,    // [4H]
    const bf16*  __restrict__ hprev,  // [B, H]
    bf16*        __restrict__ hout,   // [B, H]
    float*       __restrict__ cst)    // [B, H]
{
    __shared__ bf16 AsBuf[2][BM * APITCH];
    __shared__ bf16 BsBuf[2][BN * APITCH];
    __shared__ float smG[64 * GPITCH];

    const int tid = threadIdx.x, lane = tid & 31, wave = tid >> 5;
    const int wr = wave >> 1, wc = wave & 1;
    const int b0  = blockIdx.y * BM;
    const int hc0 = blockIdx.x * 32;

    f32x8 acc[2][4];
#pragma unroll
    for (int i = 0; i < 2; ++i)
#pragma unroll
        for (int j = 0; j < 4; ++j) acc[i][j] = f32x8_zero();

    const int arow = tid >> 1, ac0 = (tid & 1) * 16;
    const int wrow = ((arow >> 5) << 9) + hc0 + (arow & 31);   // gate*512 + hcol

    auto issue = [&](int s, int bi) {
        const bf16 *ga, *gb;
        if (s < E_ / BK) {
            const int k0 = s * BK;
            ga = embt + (size_t)(b0 + arow) * E_ + k0 + ac0;
            gb = Wih  + (size_t)wrow * E_ + k0 + ac0;
        } else {
            const int k0 = (s - E_ / BK) * BK;
            ga = hprev + (size_t)(b0 + arow) * H_ + k0 + ac0;
            gb = Whh   + (size_t)wrow * H_ + k0 + ac0;
        }
        const unsigned la = lds_addr_of(AsBuf[bi] + arow * APITCH + ac0);
        const unsigned lb = lds_addr_of(BsBuf[bi] + arow * APITCH + ac0);
        async_b128(la,      ga);
        async_b128(la + 16, (const char*)ga + 16);
        async_b128(lb,      gb);
        async_b128(lb + 16, (const char*)gb + 16);
    };

    const int NS = (E_ + H_) / BK;   // 24 stages
    issue(0, 0);
    wait_async0();
    __syncthreads();
    for (int s = 0; s < NS; ++s) {
        if (s + 1 < NS) issue(s + 1, (s + 1) & 1);
        mma_8(AsBuf[s & 1], BsBuf[s & 1], acc, wr, wc, lane);
        if (s + 1 < NS) wait_async0();
        __syncthreads();
    }

    // Epilogue: stage preactivations via LDS (two 64-row passes), apply gates.
    const int kh = lane >> 4, n16 = lane & 15;
    for (int half = 0; half < 2; ++half) {
        if ((wr >> 1) == half) {
#pragma unroll
            for (int i = 0; i < 2; ++i)
#pragma unroll
                for (int j = 0; j < 4; ++j)
#pragma unroll
                    for (int v = 0; v < 8; ++v) {
                        const int ml = wr * 32 + i * 16 + kh * 8 + v - half * 64;
                        const int nl = wc * 64 + j * 16 + n16;
                        smG[ml * GPITCH + nl] = acc[i][j][v];
                    }
        }
        __syncthreads();
        for (int idx = tid; idx < 64 * 32; idx += BDIM) {
            const int r = idx >> 5, j = idx & 31;
            const int hcol = hc0 + j;
            const float pi = smG[r * GPITCH + j]      + bih[hcol]        + bhh[hcol];
            const float pf = smG[r * GPITCH + 32 + j] + bih[512  + hcol] + bhh[512  + hcol];
            const float pg = smG[r * GPITCH + 64 + j] + bih[1024 + hcol] + bhh[1024 + hcol];
            const float po = smG[r * GPITCH + 96 + j] + bih[1536 + hcol] + bhh[1536 + hcol];
            const float si = 1.f / (1.f + __expf(-pi));
            const float sf = 1.f / (1.f + __expf(-pf));
            const float tg = tanhf(pg);
            const float so = 1.f / (1.f + __expf(-po));
            const size_t ci = (size_t)(b0 + half * 64 + r) * H_ + hcol;
            const float cn = sf * cst[ci] + si * tg;
            cst[ci]  = cn;
            hout[ci] = (bf16)(so * tanhf(cn));
        }
        __syncthreads();
    }
}

// ---- kernel 3: out[b,t,:] = h_t[b,:] @ W_agent^T + b_agent (f32 out) -------
__global__ __launch_bounds__(BDIM) void agent_kernel(
    const bf16*  __restrict__ hs,    // [T*B, H]  row m = t*B + b (slots 1..T, contiguous)
    const bf16*  __restrict__ Wag,   // [O, H]
    const float* __restrict__ bag,   // [O]
    float*       __restrict__ out)   // [B, T, O]
{
    __shared__ bf16 AsBuf[2][BM * APITCH];
    __shared__ bf16 BsBuf[2][BN * APITCH];
    const int tid = threadIdx.x, lane = tid & 31, wave = tid >> 5;
    const int wr = wave >> 1, wc = wave & 1;
    const int m0 = blockIdx.y * BM, n0 = blockIdx.x * BN;

    f32x8 acc[2][4];
#pragma unroll
    for (int i = 0; i < 2; ++i)
#pragma unroll
        for (int j = 0; j < 4; ++j) acc[i][j] = f32x8_zero();

    const int NS = H_ / BK;   // 16 stages
    async_tile(hs  + (size_t)m0 * H_, H_, AsBuf[0], tid);
    async_tile(Wag + (size_t)n0 * H_, H_, BsBuf[0], tid);
    wait_async0();
    __syncthreads();
    for (int s = 0; s < NS; ++s) {
        if (s + 1 < NS) {
            const int k0 = (s + 1) * BK;
            async_tile(hs  + (size_t)m0 * H_ + k0, H_, AsBuf[(s + 1) & 1], tid);
            async_tile(Wag + (size_t)n0 * H_ + k0, H_, BsBuf[(s + 1) & 1], tid);
        }
        mma_8(AsBuf[s & 1], BsBuf[s & 1], acc, wr, wc, lane);
        if (s + 1 < NS) wait_async0();
        __syncthreads();
    }

    const int kh = lane >> 4, n16 = lane & 15;
#pragma unroll
    for (int i = 0; i < 2; ++i)
#pragma unroll
        for (int j = 0; j < 4; ++j)
#pragma unroll
            for (int v = 0; v < 8; ++v) {
                const int m = m0 + wr * 32 + i * 16 + kh * 8 + v;
                const int n = n0 + wc * 64 + j * 16 + n16;
                const int t = m >> 9, b = m & (B_ - 1);
                out[((size_t)b * T_ + t) * O_ + n] = acc[i][j][v] + bag[n];
            }
}

// ---- small utility kernels --------------------------------------------------
__global__ void cvt_f32_to_bf16(const float* __restrict__ src, bf16* __restrict__ dst, int n)
{
    const int i = blockIdx.x * blockDim.x + threadIdx.x;
    if (i < n) dst[i] = (bf16)src[i];
}

__global__ void init_state_kernel(bf16* __restrict__ h0, float* __restrict__ c, int n)
{
    const int i = blockIdx.x * blockDim.x + threadIdx.x;
    if (i < n) { h0[i] = (bf16)0.f; c[i] = 0.f; }
}

// ---- host launcher ----------------------------------------------------------
extern "C" void kernel_launch(void* const* d_in, const int* in_sizes, int n_in,
                              void* d_out, int out_size, void* d_ws, size_t ws_size,
                              hipStream_t stream)
{
    (void)in_sizes; (void)n_in; (void)out_size; (void)ws_size;
    const float* message = (const float*)d_in[0];
    const float* W_emb   = (const float*)d_in[1];
    const float* b_emb   = (const float*)d_in[2];
    const float* W_ih    = (const float*)d_in[3];
    const float* b_ih    = (const float*)d_in[4];
    const float* W_hh    = (const float*)d_in[5];
    const float* b_hh    = (const float*)d_in[6];
    const float* W_ag    = (const float*)d_in[7];
    const float* b_ag    = (const float*)d_in[8];

    char* p = (char*)d_ws;
    bf16* Wemb_b = (bf16*)p; p += (size_t)E_  * V_ * 2;             // 0.5 MB
    bf16* Wih_b  = (bf16*)p; p += (size_t)4 * H_ * E_ * 2;          // 1 MB
    bf16* Whh_b  = (bf16*)p; p += (size_t)4 * H_ * H_ * 2;          // 2 MB
    bf16* Wag_b  = (bf16*)p; p += (size_t)O_  * H_ * 2;             // 0.25 MB
    bf16* emb    = (bf16*)p; p += (size_t)T_ * B_ * E_ * 2;         // 16.8 MB
    bf16* hs     = (bf16*)p; p += (size_t)(T_ + 1) * B_ * H_ * 2;   // 34 MB (slot 0 = h_{-1}=0)
    float* cst   = (float*)p; p += (size_t)B_ * H_ * 4;             // 1 MB

    const int nt = 256;
    cvt_f32_to_bf16<<<(E_ * V_ + nt - 1) / nt, nt, 0, stream>>>(W_emb, Wemb_b, E_ * V_);
    cvt_f32_to_bf16<<<(4 * H_ * E_ + nt - 1) / nt, nt, 0, stream>>>(W_ih, Wih_b, 4 * H_ * E_);
    cvt_f32_to_bf16<<<(4 * H_ * H_ + nt - 1) / nt, nt, 0, stream>>>(W_hh, Whh_b, 4 * H_ * H_);
    cvt_f32_to_bf16<<<(O_ * H_ + nt - 1) / nt, nt, 0, stream>>>(W_ag, Wag_b, O_ * H_);
    init_state_kernel<<<(B_ * H_ + nt - 1) / nt, nt, 0, stream>>>(hs, cst, B_ * H_);

    embed_kernel<<<dim3(E_ / BN, (B_ * T_) / BM), BDIM, 0, stream>>>(message, Wemb_b, b_emb, emb);

    for (int t = 0; t < T_; ++t) {
        lstm_step_kernel<<<dim3(H_ / 32, B_ / BM), BDIM, 0, stream>>>(
            emb + (size_t)t * B_ * E_, Wih_b, Whh_b, b_ih, b_hh,
            hs + (size_t)t * B_ * H_,            // h_{t-1}
            hs + (size_t)(t + 1) * B_ * H_,      // h_t
            cst);
    }

    agent_kernel<<<dim3(O_ / BN, (T_ * B_) / BM), BDIM, 0, stream>>>(
        hs + (size_t)B_ * H_, Wag_b, b_ag, (float*)d_out);
}
